// SA_755914244115
// MI455X (gfx1250) — compile-verified, matrix operand order checked by reading, not run on hardware
//
#include <hip/hip_runtime.h>
#include <hip/hip_bf16.h>

typedef __attribute__((ext_vector_type(16))) _Float16 v16h;
typedef __attribute__((ext_vector_type(8)))  float    v8f;

constexpr int Bb = 256, Tt = 128, Nn = 512, Hh = 128, Gg = 4 * Hh; // Gg = 512
constexpr int ROWS = 16;          // batch rows per workgroup == WMMA M
constexpr int THREADS = 512;      // 16 wave32s

__device__ __forceinline__ float sigf(float x) { return 1.0f / (1.0f + __expf(-x)); }

// ---- one-time fp32 -> fp16 weight conversion into workspace ----
__global__ void cvt_weights(const float* __restrict__ w_ih, const float* __restrict__ w_hh,
                            _Float16* __restrict__ wih_h, _Float16* __restrict__ whh_h) {
  int idx = blockIdx.x * blockDim.x + threadIdx.x;
  const int n_ih = Gg * Nn, n_hh = Gg * Hh;
  if (idx < n_ih)               wih_h[idx]        = (_Float16)w_ih[idx];
  else if (idx < n_ih + n_hh)   whh_h[idx - n_ih] = (_Float16)w_hh[idx - n_ih];
}

// ---- persistent attention-LSTM: one block = 16 batch rows, all T steps ----
__global__ __launch_bounds__(THREADS)
void attn_lstm_persistent(const float* __restrict__ X,
                          const float* __restrict__ attn_w,
                          const float* __restrict__ attn_b_p,
                          const float* __restrict__ b_ih,
                          const float* __restrict__ b_hh,
                          const _Float16* __restrict__ wih_h,
                          const _Float16* __restrict__ whh_h,
                          float* __restrict__ out_xt,
                          float* __restrict__ out_xe) {
  extern __shared__ char smem[];
  float*    xscore = (float*)smem;                    // ROWS*Nn  (32 KB)
  float*    scr    = xscore + ROWS * Nn;              // ROWS*Nn  (32 KB) exp scratch, then gates
  float*    hbuf   = scr + ROWS * Nn;                 // ROWS*Hh  ( 8 KB)
  float*    cbuf   = hbuf + ROWS * Hh;                // ROWS*Hh  ( 8 KB)
  _Float16* xth    = (_Float16*)(cbuf + ROWS * Hh);   // ROWS*Nn  (16 KB) x_tilde f16
  _Float16* hh     = xth + ROWS * Nn;                 // ROWS*Hh  ( 4 KB) h f16

  const int tid  = threadIdx.x;
  const int lane = tid & 31;
  const int wv   = tid >> 5;        // 0..15  (one wave per batch row for softmax)
  const int l16  = lane & 15;
  const int hi   = lane >> 4;       // 0/1 half-wave
  const int r0   = blockIdx.x * ROWS;

  const float attn_b = attn_b_p[0];
  const float* w_h = attn_w;
  const float* w_s = attn_w + Hh;
  const float* w_x = attn_w + 2 * Hh;

  // init h, c
  for (int i = tid; i < ROWS * Hh; i += THREADS) {
    hbuf[i] = 0.f; cbuf[i] = 0.f; hh[i] = (_Float16)0.f;
  }

  // x_score[r][n] = sum_t X[b, t, n] * w_x[t]
  for (int e = tid; e < ROWS * Nn; e += THREADS) {
    int r = e >> 9, n = e & (Nn - 1);
    const float* xp = X + (size_t)(r0 + r) * Tt * Nn + n;
    float acc = 0.f;
    #pragma unroll 4
    for (int t = 0; t < Tt; ++t) acc += xp[(size_t)t * Nn] * w_x[t];
    xscore[e] = acc;
  }
  __syncthreads();

  for (int t = 0; t < Tt; ++t) {
    // ---- Phase B: per-row attn scalar + softmax over N + x_tilde (one row/wave) ----
    {
      int r = wv;
      // attention scalar for this wave's row: h.w_h + c.w_s + b (32-lane reduce)
      float p = 0.f;
      #pragma unroll
      for (int j = 0; j < Hh / 32; ++j) {
        int k = lane + 32 * j;
        p += hbuf[r * Hh + k] * w_h[k] + cbuf[r * Hh + k] * w_s[k];
      }
      #pragma unroll
      for (int off = 16; off >= 1; off >>= 1) p += __shfl_xor(p, off, 32);
      float s = p + attn_b;

      float m = -3.0e38f;
      #pragma unroll 8
      for (int i = 0; i < Nn / 32; ++i)
        m = fmaxf(m, xscore[r * Nn + lane + i * 32] + s);
      #pragma unroll
      for (int off = 16; off >= 1; off >>= 1) m = fmaxf(m, __shfl_xor(m, off, 32));

      float sum = 0.f;
      #pragma unroll 8
      for (int i = 0; i < Nn / 32; ++i) {
        int n = lane + i * 32;
        float e = __expf(xscore[r * Nn + n] + s - m);
        scr[r * Nn + n] = e;
        sum += e;
      }
      #pragma unroll
      for (int off = 16; off >= 1; off >>= 1) sum += __shfl_xor(sum, off, 32);
      float inv = 1.0f / sum;

      const float* xrow = X      + (size_t)(r0 + r) * Tt * Nn + (size_t)t * Nn;
      float*       orow = out_xt + (size_t)(r0 + r) * Tt * Nn + (size_t)t * Nn;
      #pragma unroll 8
      for (int i = 0; i < Nn / 32; ++i) {
        int n = lane + i * 32;
        float xt = scr[r * Nn + n] * inv * xrow[n];
        orow[n] = xt;
        xth[r * Nn + n] = (_Float16)xt;
      }
      // warm the cache for the next timestep's x_t while WMMAs run
      if (t + 1 < Tt) {
        #pragma unroll
        for (int i = 0; i < Nn / 32; i += 8)
          __builtin_prefetch(xrow + Nn + lane + i * 32, 0, 0);
      }
    }
    __syncthreads();

    // ---- Phase C: gates = x_tilde @ w_ih^T + h @ w_hh^T + bias ----
    // Each wave owns two 16x16 N-tiles; both accumulated concurrently so the
    // two v_wmma chains are independent (pipelined) and share one A fragment.
    {
      int nc0 = (wv * 2 + 0) * 16 + l16;
      int nc1 = (wv * 2 + 1) * 16 + l16;
      v8f acc0 = {}, acc1 = {};
      for (int kt = 0; kt < Nn / 32; ++kt) {          // K over x_tilde (512)
        v16h a, b0, b1;
        #pragma unroll
        for (int v = 0; v < 8; ++v) {
          int kb = ((v < 4) ? 2 * v : 16 + 2 * (v - 4)) + hi * 8;
          int k  = kt * 32 + kb;
          a [2 * v]     = xth[l16 * Nn + k];
          a [2 * v + 1] = xth[l16 * Nn + k + 1];
          b0[2 * v]     = wih_h[(size_t)nc0 * Nn + k];
          b0[2 * v + 1] = wih_h[(size_t)nc0 * Nn + k + 1];
          b1[2 * v]     = wih_h[(size_t)nc1 * Nn + k];
          b1[2 * v + 1] = wih_h[(size_t)nc1 * Nn + k + 1];
        }
        acc0 = __builtin_amdgcn_wmma_f32_16x16x32_f16(false, a, false, b0,
                                                      (short)0, acc0, false, false);
        acc1 = __builtin_amdgcn_wmma_f32_16x16x32_f16(false, a, false, b1,
                                                      (short)0, acc1, false, false);
      }
      for (int kt = 0; kt < Hh / 32; ++kt) {          // K over h (128)
        v16h a, b0, b1;
        #pragma unroll
        for (int v = 0; v < 8; ++v) {
          int kb = ((v < 4) ? 2 * v : 16 + 2 * (v - 4)) + hi * 8;
          int k  = kt * 32 + kb;
          a [2 * v]     = hh[l16 * Hh + k];
          a [2 * v + 1] = hh[l16 * Hh + k + 1];
          b0[2 * v]     = whh_h[(size_t)nc0 * Hh + k];
          b0[2 * v + 1] = whh_h[(size_t)nc0 * Hh + k + 1];
          b1[2 * v]     = whh_h[(size_t)nc1 * Hh + k];
          b1[2 * v + 1] = whh_h[(size_t)nc1 * Hh + k + 1];
        }
        acc0 = __builtin_amdgcn_wmma_f32_16x16x32_f16(false, a, false, b0,
                                                      (short)0, acc0, false, false);
        acc1 = __builtin_amdgcn_wmma_f32_16x16x32_f16(false, a, false, b1,
                                                      (short)0, acc1, false, false);
      }
      float bias0 = b_ih[nc0] + b_hh[nc0];
      float bias1 = b_ih[nc1] + b_hh[nc1];
      #pragma unroll
      for (int p = 0; p < 8; ++p) {
        int m = p + 8 * hi;             // C/D layout: lanes 0-15 -> M=p, 16-31 -> M=p+8
        scr[m * Nn + nc0] = acc0[p] + bias0;
        scr[m * Nn + nc1] = acc1[p] + bias1;
      }
    }
    __syncthreads();

    // ---- Phase D: LSTM cell update ----
    for (int e = tid; e < ROWS * Hh; e += THREADS) {
      int r = e >> 7, k = e & (Hh - 1);
      float ig = scr[r * Nn + k];
      float fg = scr[r * Nn + Hh + k];
      float gg = scr[r * Nn + 2 * Hh + k];
      float og = scr[r * Nn + 3 * Hh + k];
      float co = cbuf[e];
      float cn = sigf(fg) * co + sigf(ig) * tanhf(gg);
      float hn = sigf(og) * tanhf(cn);
      cbuf[e] = cn;
      hbuf[e] = hn;
      hh[e]   = (_Float16)hn;
      out_xe[(size_t)(r0 + r) * Tt * Hh + (size_t)t * Hh + k] = hn;
    }
    __syncthreads();
  }
}

extern "C" void kernel_launch(void* const* d_in, const int* in_sizes, int n_in,
                              void* d_out, int out_size, void* d_ws, size_t ws_size,
                              hipStream_t stream) {
  const float* X      = (const float*)d_in[0];
  const float* attn_w = (const float*)d_in[1];
  const float* attn_b = (const float*)d_in[2];
  const float* w_ih   = (const float*)d_in[3];
  const float* w_hh   = (const float*)d_in[4];
  const float* b_ih   = (const float*)d_in[5];
  const float* b_hh   = (const float*)d_in[6];

  float* out_xt = (float*)d_out;
  float* out_xe = out_xt + (size_t)Bb * Tt * Nn;

  _Float16* wih_h = (_Float16*)d_ws;
  _Float16* whh_h = wih_h + (size_t)Gg * Nn;

  int ncvt = Gg * Nn + Gg * Hh;
  cvt_weights<<<(ncvt + 255) / 256, 256, 0, stream>>>(w_ih, w_hh, wih_h, whh_h);

  size_t smem = (size_t)(2 * ROWS * Nn + 2 * ROWS * Hh) * sizeof(float)   // xscore, scr, h, c
              + (size_t)(ROWS * Nn + ROWS * Hh) * sizeof(_Float16);       // xth, hh
  attn_lstm_persistent<<<Bb / ROWS, THREADS, smem, stream>>>(
      X, attn_w, attn_b, b_ih, b_hh, wih_h, whh_h, out_xt, out_xe);
}
